// _AdaPVConv_5033701671430
// MI455X (gfx1250) — matrix operand order repaired and checked by reading, not compiled, not run on hardware
//
#include <hip/hip_runtime.h>
#include <hip/hip_bf16.h>
#include <cstdint>
#include <cstddef>

// Problem constants (match reference)
#define BATCH 4
#define CINN  32
#define CONN  64      // COUT
#define NPT   32768   // points
#define RR    32      // voxel resolution
#define NVOX  32768   // R^3  (== NPT, convenient)
#define SDIM  256

typedef __attribute__((ext_vector_type(16))) __bf16        v16bf;
typedef __attribute__((ext_vector_type(8)))  float         v8f;
typedef __attribute__((ext_vector_type(8)))  unsigned int  v8u;
typedef __attribute__((ext_vector_type(4)))  unsigned int  u32x4;
typedef __attribute__((ext_vector_type(8)))  int           i32x8_t;
typedef __attribute__((ext_vector_type(4)))  int           i32x4_t;

union BF16x16 { v8u u; v16bf h; };

__device__ __forceinline__ unsigned short f2bf(float f) {
  unsigned int u = __float_as_uint(f);
  u += 0x7FFFu + ((u >> 16) & 1u);   // round-to-nearest-even
  return (unsigned short)(u >> 16);
}
__device__ __forceinline__ float silu_f(float x) {
  return x / (1.0f + __expf(-x));
}

// ---------------------------------------------------------------------------
// Tensor Data Mover: 1D global->LDS async copy of nelem f32 elements.
// D# built per CDNA5 ISA ch.8: group0 = {count=1, lds_addr, global_addr, type=2},
// group1 = {data_size=4B, tensor_dim0=nelem, tensor_dim1=1, tile_dim0=nelem,
//           tensor_dim0_stride=nelem}. Tracked by TENSORcnt.
// ---------------------------------------------------------------------------
__device__ __forceinline__ void tdm_load_1d_f32(const void* gsrc, unsigned lds_addr,
                                                unsigned nelem) {
  unsigned long long ga = (unsigned long long)(uintptr_t)gsrc;
  u32x4 g0;
  g0[0] = 1u;                                                  // count=1 (valid)
  g0[1] = lds_addr;                                            // bits 63:32
  g0[2] = (unsigned)ga;                                        // global_addr lo
  g0[3] = (unsigned)((ga >> 32) & 0x01FFFFFFull) | (2u << 30); // addr hi | type=2
  i32x8_t g1;
  g1[0] = (int)(2u << 16);                                     // data_size = 4 bytes
  g1[1] = (int)((nelem & 0xFFFFu) << 16);                      // tensor_dim0[15:0]
  g1[2] = (int)(((nelem >> 16) & 0xFFFFu) | (1u << 16));       // dim0[31:16] | dim1=1
  g1[3] = (int)((nelem & 0xFFFFu) << 16);                      // tile_dim0
  g1[4] = 0;                                                   // tile_dim1/2 unused
  g1[5] = (int)nelem;                                          // dim0_stride lo32
  g1[6] = 0;
  g1[7] = 0;
  i32x4_t gz = {0, 0, 0, 0};
#if defined(__clang_major__) && (__clang_major__ >= 23)
  i32x8_t gz8 = {0, 0, 0, 0, 0, 0, 0, 0};
  __builtin_amdgcn_tensor_load_to_lds(g0, g1, gz, gz, gz8, 0);
#else
  __builtin_amdgcn_tensor_load_to_lds(g0, g1, gz, gz, 0);
#endif
}

// ---------------------------------------------------------------------------
// Weight repack into WMMA bf16 A-matrix lane layout.
// Contiguous-k A order used here: lane<16: M=m0+lane, K=j
//                        lane>=16: M=m0+lane-16, K=16+j
// Apack index = (((kc*4)+mt)*32 + lane)*16 + j
// ---------------------------------------------------------------------------
__global__ void k_pack_w(const float* __restrict__ W, unsigned short* __restrict__ Apack, int K) {
  int total = (K / 32) * 4 * 32 * 16;
  for (int idx = blockIdx.x * blockDim.x + threadIdx.x; idx < total; idx += gridDim.x * blockDim.x) {
    int j    = idx & 15;
    int lane = (idx >> 4) & 31;
    int mt   = (idx >> 9) & 3;
    int kc   = idx >> 11;
    int o     = mt * 16 + (lane & 15);
    int khalf = (lane < 16) ? 0 : 16;
    int k     = kc * 32 + khalf + j;
    Apack[idx] = f2bf(W[o * K + k]);
  }
}

// conv3d weights W[o][i][3][3][3] -> Apack[t][kc][mt][lane][j], t = dx*9+dy*3+dz
__global__ void k_pack_wconv(const float* __restrict__ W, unsigned short* __restrict__ Apack) {
  int total = 27 * 2 * 4 * 32 * 16;
  for (int idx = blockIdx.x * blockDim.x + threadIdx.x; idx < total; idx += gridDim.x * blockDim.x) {
    int j    = idx & 15;
    int lane = (idx >> 4) & 31;
    int mt   = (idx >> 9) & 3;
    int kc   = (idx >> 11) & 1;
    int t    = idx >> 12;
    int o     = mt * 16 + (lane & 15);
    int khalf = (lane < 16) ? 0 : 16;
    int i     = kc * 32 + khalf + j;
    Apack[idx] = f2bf(W[(o * CONN + i) * 27 + t]);
  }
}

__global__ void k_f2bf(const float* __restrict__ X, unsigned short* __restrict__ Y, int n) {
  for (int i = blockIdx.x * blockDim.x + threadIdx.x; i < n; i += gridDim.x * blockDim.x)
    Y[i] = f2bf(X[i]);
}

// ---------------------------------------------------------------------------
// Per-batch bbox of coords (B,N,3)
// ---------------------------------------------------------------------------
__global__ void k_bbox(const float* __restrict__ coords, float* __restrict__ bbox) {
  const int b = blockIdx.x;
  const float* c = coords + (size_t)b * NPT * 3;
  float mn[3] = {1e30f, 1e30f, 1e30f}, mx[3] = {-1e30f, -1e30f, -1e30f};
  for (int n = threadIdx.x; n < NPT; n += 256) {
    #pragma unroll
    for (int d = 0; d < 3; ++d) {
      float v = c[n * 3 + d];
      mn[d] = fminf(mn[d], v);
      mx[d] = fmaxf(mx[d], v);
    }
  }
  __shared__ float smn[3][256], smx[3][256];
  #pragma unroll
  for (int d = 0; d < 3; ++d) { smn[d][threadIdx.x] = mn[d]; smx[d][threadIdx.x] = mx[d]; }
  __syncthreads();
  for (int s = 128; s > 0; s >>= 1) {
    if (threadIdx.x < s) {
      #pragma unroll
      for (int d = 0; d < 3; ++d) {
        smn[d][threadIdx.x] = fminf(smn[d][threadIdx.x], smn[d][threadIdx.x + s]);
        smx[d][threadIdx.x] = fmaxf(smx[d][threadIdx.x], smx[d][threadIdx.x + s]);
      }
    }
    __syncthreads();
  }
  if (threadIdx.x == 0) {
    #pragma unroll
    for (int d = 0; d < 3; ++d) { bbox[b * 6 + d] = smn[d][0]; bbox[b * 6 + 3 + d] = smx[d][0]; }
  }
}

// ---------------------------------------------------------------------------
// GEMM: Out[b][m][n] = sum_k Apack(m,k) * Xbf[b][k][n]   (M=64, N tile=32/block)
// 8 waves: 4 M-tiles x 2 N-subtiles; WMMA f32_16x16x32_bf16 with f32 accum.
// ---------------------------------------------------------------------------
__global__ void k_gemm(const unsigned short* __restrict__ Apack,
                       const unsigned short* __restrict__ Xbf,
                       float* __restrict__ Out, int K) {
  const int nt   = blockIdx.x & 1023;       // NPT/32 = 1024
  const int b    = blockIdx.x >> 10;
  const int n0   = nt * 32;
  const int lane = threadIdx.x & 31;
  const int wave = threadIdx.x >> 5;
  const int mt   = wave & 3;
  const int nsub = (wave >> 2) << 4;

  __shared__ unsigned short Xs[32][66];     // [n_local][k], row pad 66 -> no bank conflicts

  for (int e = threadIdx.x; e < 32 * K; e += 256) {
    int k = e >> 5, nl = e & 31;
    Xs[nl][k] = Xbf[((size_t)b * K + k) * NPT + n0 + nl];
  }
  __syncthreads();

  v8f acc = {};
  const int nl    = nsub + (lane & 15);
  const int khalf = (lane < 16) ? 0 : 16;
  for (int kc = 0; kc < (K >> 5); ++kc) {
    BF16x16 a, bb;
    a.u = *(const v8u*)(Apack + (((size_t)kc * 4 + mt) * 32 + lane) * 16);
    const unsigned int* xr = (const unsigned int*)&Xs[nl][kc * 32 + khalf];
    #pragma unroll
    for (int q = 0; q < 8; ++q) bb.u[q] = xr[q];
    acc = __builtin_amdgcn_wmma_f32_16x16x32_bf16(false, a.h, false, bb.h, (short)0, acc, false, false);
  }
  const int mrow = mt * 16 + ((lane < 16) ? 0 : 8);
  #pragma unroll
  for (int r = 0; r < 8; ++r)
    Out[((size_t)b * CONN + mrow + r) * NPT + n0 + nl] = acc[r];
}

// ---------------------------------------------------------------------------
// 3x3x3 conv via implicit GEMM: block = one (x,y) z-row (32 voxels), M=64,
// K = 27 taps x 64 ch. LDS tile [z+halo=34][ch=64 pad 66] bf16.
// ---------------------------------------------------------------------------
__global__ void k_conv3d(const unsigned short* __restrict__ Apack,   // [27][2][4][32][16]
                         const unsigned short* __restrict__ Xbf,     // [B][64][32768]
                         float* __restrict__ Out) {
  const int row  = blockIdx.x & 1023;   // x*32+y
  const int b    = blockIdx.x >> 10;
  const int x    = row >> 5, y = row & 31;
  const int lane = threadIdx.x & 31;
  const int wave = threadIdx.x >> 5;
  const int mt   = wave & 3;
  const int nsub = (wave >> 2) << 4;

  __shared__ unsigned short Xs[34][66];
  // zero z-halo rows once (staging only writes rows 1..32)
  if (threadIdx.x < 66)       Xs[0][threadIdx.x] = 0;
  else if (threadIdx.x < 132) Xs[33][threadIdx.x - 66] = 0;

  v8f acc = {};
  const int zl    = nsub + (lane & 15);        // output z of this lane
  const int khalf = (lane < 16) ? 0 : 16;
  const int mrow  = mt * 16 + ((lane < 16) ? 0 : 8);
  const int ci    = threadIdx.x >> 2;          // staging channel 0..63
  const int qz    = (threadIdx.x & 3) * 8;     // staging z block

  for (int tap9 = 0; tap9 < 9; ++tap9) {
    const int dx = tap9 / 3 - 1, dy = tap9 % 3 - 1;
    __syncthreads();                           // previous tap fully consumed
    const int xx = x + dx, yy = y + dy;
    uint4 v = make_uint4(0u, 0u, 0u, 0u);
    if (xx >= 0 && xx < RR && yy >= 0 && yy < RR) {
      v = *(const uint4*)(Xbf + ((size_t)b * CONN + ci) * NVOX + (size_t)((xx << 5) + yy) * 32 + qz);
    }
    Xs[1 + qz + 0][ci] = (unsigned short)(v.x);
    Xs[1 + qz + 1][ci] = (unsigned short)(v.x >> 16);
    Xs[1 + qz + 2][ci] = (unsigned short)(v.y);
    Xs[1 + qz + 3][ci] = (unsigned short)(v.y >> 16);
    Xs[1 + qz + 4][ci] = (unsigned short)(v.z);
    Xs[1 + qz + 5][ci] = (unsigned short)(v.z >> 16);
    Xs[1 + qz + 6][ci] = (unsigned short)(v.w);
    Xs[1 + qz + 7][ci] = (unsigned short)(v.w >> 16);
    // prefetch next tap's row into cache (global_prefetch_b8)
    if (tap9 + 1 < 9) {
      const int nx = x + (tap9 + 1) / 3 - 1, ny = y + (tap9 + 1) % 3 - 1;
      if (nx >= 0 && nx < RR && ny >= 0 && ny < RR)
        __builtin_prefetch(Xbf + ((size_t)b * CONN + ci) * NVOX + (size_t)((nx << 5) + ny) * 32 + qz, 0, 3);
    }
    __syncthreads();

    #pragma unroll
    for (int dz = 0; dz < 3; ++dz) {
      const int t = tap9 * 3 + dz;
      #pragma unroll
      for (int kc = 0; kc < 2; ++kc) {
        BF16x16 a, bb;
        a.u = *(const v8u*)(Apack + (((size_t)t * 2 + kc) * 4 + mt) * 512 + lane * 16);
        const unsigned int* xr = (const unsigned int*)&Xs[zl + dz][kc * 32 + khalf];
        #pragma unroll
        for (int q = 0; q < 8; ++q) bb.u[q] = xr[q];
        acc = __builtin_amdgcn_wmma_f32_16x16x32_bf16(false, a.h, false, bb.h, (short)0, acc, false, false);
      }
    }
  }
  #pragma unroll
  for (int r = 0; r < 8; ++r)
    Out[((size_t)b * CONN + mrow + r) * NVOX + (size_t)row * 32 + zl] = acc[r];
}

// ---------------------------------------------------------------------------
// Per-(b,c) sum / sumsq over N for GroupNorm, fed by TDM async global->LDS
// double buffering: wave 0 issues tensor_load_to_lds for chunk i+1 while the
// whole block reduces chunk i from LDS. Drain with s_wait_tensorcnt 0.
// ---------------------------------------------------------------------------
#define STAT_CHUNK 4096   // floats per chunk (16 KB); NPT/STAT_CHUNK = 8 chunks

__global__ void k_stats(const float* __restrict__ X, float* __restrict__ sums, float* __restrict__ sqs) {
  const int bc = blockIdx.x;
  const float* p = X + (size_t)bc * NPT;
  __shared__ float buf[2][STAT_CHUNK];
  __shared__ float red[512];
  const int tid = threadIdx.x;

  if (tid < 32)   // one TDM issue per block (per-wave op, EXEC ignored)
    tdm_load_1d_f32(p, (unsigned)(uintptr_t)&buf[0][0], STAT_CHUNK);

  float s = 0.f, q = 0.f;
  for (int i = 0; i < NPT / STAT_CHUNK; ++i) {
    if (tid < 32)
      __builtin_amdgcn_s_wait_tensorcnt(0);    // chunk i landed in LDS
    __syncthreads();
    if (tid < 32 && (i + 1) < NPT / STAT_CHUNK)
      tdm_load_1d_f32(p + (size_t)(i + 1) * STAT_CHUNK,
                      (unsigned)(uintptr_t)&buf[(i + 1) & 1][0], STAT_CHUNK);
    const float* cb = &buf[i & 1][0];
    for (int j = tid; j < STAT_CHUNK; j += 256) { float v = cb[j]; s += v; q += v * v; }
    __syncthreads();                           // done reading buf[i&1] before reuse
  }

  red[tid] = s; red[256 + tid] = q;
  __syncthreads();
  for (int st = 128; st > 0; st >>= 1) {
    if (tid < st) { red[tid] += red[tid + st]; red[256 + tid] += red[256 + tid + st]; }
    __syncthreads();
  }
  if (tid == 0) { sums[bc] = red[0]; sqs[bc] = red[256]; }
}

__device__ __forceinline__ void group_stats(const float* sums, const float* sqs,
                                            int b, int g, float& mean, float& rstd) {
  float s = 0.f, q = 0.f;
  #pragma unroll
  for (int c = 0; c < 8; ++c) { s += sums[b * CONN + g * 8 + c]; q += sqs[b * CONN + g * 8 + c]; }
  const float inv = 1.0f / (8.0f * (float)NPT);
  mean = s * inv;
  float var = q * inv - mean * mean;
  rstd = rsqrtf(var + 1e-5f);
}

// ---------------------------------------------------------------------------
// GN + SiLU + average-voxelize scatter (atomics)
// ---------------------------------------------------------------------------
__global__ void k_gn_vox(const float* __restrict__ Y, const float* __restrict__ sums,
                         const float* __restrict__ sqs,
                         const float* __restrict__ gamma, const float* __restrict__ beta,
                         const float* __restrict__ coords, const float* __restrict__ bbox,
                         float* __restrict__ voxsum, float* __restrict__ cnt) {
  const int chunk = blockIdx.x & 127;
  const int b     = blockIdx.x >> 7;
  const int n     = chunk * 256 + threadIdx.x;
  __shared__ float gm[8], gr[8];
  if (threadIdx.x < 8) { float m, r; group_stats(sums, sqs, b, threadIdx.x, m, r); gm[threadIdx.x] = m; gr[threadIdx.x] = r; }
  __syncthreads();

  const float* cc = coords + ((size_t)b * NPT + n) * 3;
  const float* bx = bbox + b * 6;
  int vi[3];
  #pragma unroll
  for (int d = 0; d < 3; ++d) {
    float f = (cc[d] - bx[d]) / fmaxf(bx[3 + d] - bx[d], 1e-8f) * (float)(RR - 1);
    int ii = (int)rintf(f);
    vi[d] = min(max(ii, 0), RR - 1);
  }
  const int v = (vi[0] * RR + vi[1]) * RR + vi[2];
  atomicAdd(&cnt[b * NVOX + v], 1.0f);
  for (int c = 0; c < CONN; ++c) {
    float val = Y[((size_t)b * CONN + c) * NPT + n];
    const int g = c >> 3;
    val = (val - gm[g]) * gr[g] * gamma[c] + beta[c];
    val = silu_f(val);
    atomicAdd(&voxsum[((size_t)b * CONN + c) * NVOX + v], val);
  }
}

__global__ void k_voxnorm(const float* __restrict__ voxsum, const float* __restrict__ cnt,
                          unsigned short* __restrict__ out) {
  const size_t i = (size_t)blockIdx.x * 256 + threadIdx.x;  // B*64*NVOX total
  const int v = (int)(i & (NVOX - 1));
  const int b = (int)(i >> 21);                             // 64*NVOX = 2^21
  const float c = cnt[b * NVOX + v];
  out[i] = f2bf(voxsum[i] / fmaxf(c, 1.0f));
}

// GN + SiLU -> bf16 (out_bf) or f32 (out_f)
__global__ void k_gn_act(const float* __restrict__ Y, const float* __restrict__ sums,
                         const float* __restrict__ sqs,
                         const float* __restrict__ gamma, const float* __restrict__ beta,
                         unsigned short* __restrict__ out_bf, float* __restrict__ out_f) {
  const int chunk = blockIdx.x & 127;
  const int b     = blockIdx.x >> 7;
  const int n     = chunk * 256 + threadIdx.x;
  __shared__ float gm[8], gr[8];
  if (threadIdx.x < 8) { float m, r; group_stats(sums, sqs, b, threadIdx.x, m, r); gm[threadIdx.x] = m; gr[threadIdx.x] = r; }
  __syncthreads();
  for (int c = 0; c < CONN; ++c) {
    const size_t idx = ((size_t)b * CONN + c) * NPT + n;
    float val = Y[idx];
    const int g = c >> 3;
    val = silu_f((val - gm[g]) * gr[g] * gamma[c] + beta[c]);
    if (out_bf) out_bf[idx] = f2bf(val);
    else        out_f[idx]  = val;
  }
}

// ---------------------------------------------------------------------------
// Trilinear devoxelize gather -> bf16 (feeds fuse GEMM)
// ---------------------------------------------------------------------------
__global__ void k_devox(const float* __restrict__ vox, const float* __restrict__ coords,
                        const float* __restrict__ bbox, unsigned short* __restrict__ out_bf) {
  const int chunk = blockIdx.x & 127;
  const int b     = blockIdx.x >> 7;
  const int n     = chunk * 256 + threadIdx.x;
  const float* cc = coords + ((size_t)b * NPT + n) * 3;
  const float* bx = bbox + b * 6;
  float fr[3]; int x0[3];
  #pragma unroll
  for (int d = 0; d < 3; ++d) {
    float f = (cc[d] - bx[d]) / fmaxf(bx[3 + d] - bx[d], 1e-8f) * (float)(RR - 1);
    f = fminf(fmaxf(f, 0.0f), (float)(RR - 1));
    float fl = floorf(f);
    x0[d] = (int)fl;
    fr[d] = f - fl;
  }
  int idx8[8]; float w8[8];
  #pragma unroll
  for (int k = 0; k < 8; ++k) {
    const int d0 = (k >> 2) & 1, d1 = (k >> 1) & 1, d2 = k & 1;
    const int c0 = min(x0[0] + d0, RR - 1);
    const int c1 = min(x0[1] + d1, RR - 1);
    const int c2 = min(x0[2] + d2, RR - 1);
    idx8[k] = (c0 * RR + c1) * RR + c2;
    w8[k] = (d0 ? fr[0] : 1.f - fr[0]) * (d1 ? fr[1] : 1.f - fr[1]) * (d2 ? fr[2] : 1.f - fr[2]);
  }
  for (int c = 0; c < CONN; ++c) {
    const float* vf = vox + ((size_t)b * CONN + c) * NVOX;
    float s = 0.f;
    #pragma unroll
    for (int k = 0; k < 8; ++k) s += vf[idx8[k]] * w8[k];
    out_bf[((size_t)b * CONN + c) * NPT + n] = f2bf(s);
  }
}

// AdaIN scale/bias: sbuf[b][o] = style[b] . w_as[o] + b_as[o]  (one 256-thread block)
__global__ void k_adain(const float* __restrict__ style,
                        const float* __restrict__ w_as, const float* __restrict__ b_as,
                        const float* __restrict__ w_ab, const float* __restrict__ b_ab,
                        float* __restrict__ sbuf, float* __restrict__ bbuf) {
  const int i = threadIdx.x;          // b*64 + o
  const int b = i >> 6, o = i & 63;
  const float* st = style + b * SDIM;
  float s = 0.f, bb = 0.f;
  for (int d = 0; d < SDIM; ++d) {
    const float sv = st[d];
    s  += sv * w_as[o * SDIM + d];
    bb += sv * w_ab[o * SDIM + d];
  }
  sbuf[i] = s + b_as[o];
  bbuf[i] = bb + b_ab[o];
}

// final: silu(gn(fuse)) + skip, then AdaIN
__global__ void k_final(const float* __restrict__ Yfuse, const float* __restrict__ sums,
                        const float* __restrict__ sqs,
                        const float* __restrict__ gamma, const float* __restrict__ beta,
                        const float* __restrict__ skip,
                        const float* __restrict__ sbuf, const float* __restrict__ bbuf,
                        float* __restrict__ out) {
  const int chunk = blockIdx.x & 127;
  const int b     = blockIdx.x >> 7;
  const int n     = chunk * 256 + threadIdx.x;
  __shared__ float gm[8], gr[8];
  if (threadIdx.x < 8) { float m, r; group_stats(sums, sqs, b, threadIdx.x, m, r); gm[threadIdx.x] = m; gr[threadIdx.x] = r; }
  __syncthreads();
  for (int c = 0; c < CONN; ++c) {
    const size_t idx = ((size_t)b * CONN + c) * NPT + n;
    float val = Yfuse[idx];
    const int g = c >> 3;
    val = silu_f((val - gm[g]) * gr[g] * gamma[c] + beta[c]);
    val += skip[idx];
    out[idx] = val * (1.0f + sbuf[b * CONN + c]) + bbuf[b * CONN + c];
  }
}

// ---------------------------------------------------------------------------
extern "C" void kernel_launch(void* const* d_in, const int* in_sizes, int n_in,
                              void* d_out, int out_size, void* d_ws, size_t ws_size,
                              hipStream_t stream) {
  const float* feats  = (const float*)d_in[0];
  const float* coords = (const float*)d_in[1];
  const float* style  = (const float*)d_in[2];
  const float* w_pin  = (const float*)d_in[3];
  const float* g_pin  = (const float*)d_in[4];
  const float* b_pin  = (const float*)d_in[5];
  const float* w_v1   = (const float*)d_in[6];
  const float* g_v1   = (const float*)d_in[7];
  const float* b_v1   = (const float*)d_in[8];
  const float* w_v2   = (const float*)d_in[9];
  const float* g_v2   = (const float*)d_in[10];
  const float* b_v2   = (const float*)d_in[11];
  const float* w_fuse = (const float*)d_in[12];
  const float* g_f    = (const float*)d_in[13];
  const float* b_f    = (const float*)d_in[14];
  const float* w_skip = (const float*)d_in[15];
  const float* w_as   = (const float*)d_in[16];
  const float* b_as   = (const float*)d_in[17];
  const float* w_ab   = (const float*)d_in[18];
  const float* b_ab   = (const float*)d_in[19];

  char* ws = (char*)d_ws;
  const size_t MB = 1024ull * 1024ull;
  float*          F0   = (float*)(ws + 0);          // 32 MB
  float*          F1   = (float*)(ws + 32 * MB);    // 32 MB
  float*          F2   = (float*)(ws + 64 * MB);    // 32 MB
  unsigned short* H0   = (unsigned short*)(ws + 96 * MB);   // 16 MB bf16
  unsigned short* H1   = (unsigned short*)(ws + 112 * MB);  // 16 MB bf16
  unsigned short* Hf   = (unsigned short*)(ws + 128 * MB);  // 8 MB bf16 (feats)
  char* S = ws + 136 * MB;
  unsigned short* Apin  = (unsigned short*)(S + 0);
  unsigned short* Askip = (unsigned short*)(S + 4096);
  unsigned short* Afuse = (unsigned short*)(S + 8192);
  unsigned short* Av1   = (unsigned short*)(S + 16384);
  unsigned short* Av2   = (unsigned short*)(S + 16384 + 221184);
  float* cnt  = (float*)(S + 458752);
  float* bbox = (float*)(S + 983040);
  float* sums = (float*)(S + 983296);
  float* sqs  = (float*)(S + 984320);
  float* sbuf = (float*)(S + 985344);
  float* bbuf = (float*)(S + 986368);

  // Prep: bf16 conversions and weight repacks (tiny)
  k_f2bf<<<4096, 256, 0, stream>>>(feats, Hf, BATCH * CINN * NPT);
  k_pack_w<<<8, 256, 0, stream>>>(w_pin,  Apin,  CINN);
  k_pack_w<<<8, 256, 0, stream>>>(w_skip, Askip, CINN);
  k_pack_w<<<16, 256, 0, stream>>>(w_fuse, Afuse, CONN);
  k_pack_wconv<<<432, 256, 0, stream>>>(w_v1, Av1);
  k_pack_wconv<<<432, 256, 0, stream>>>(w_v2, Av2);
  k_bbox<<<BATCH, 256, 0, stream>>>(coords, bbox);

  // point_in conv1x1 (WMMA)
  k_gemm<<<BATCH * 1024, 256, 0, stream>>>(Apin, Hf, F0, CINN);
  k_stats<<<BATCH * CONN, 256, 0, stream>>>(F0, sums, sqs);

  // voxelize (GN+SiLU fused into scatter)
  hipMemsetAsync(F1, 0, (size_t)BATCH * CONN * NVOX * sizeof(float), stream);
  hipMemsetAsync(cnt, 0, (size_t)BATCH * NVOX * sizeof(float), stream);
  k_gn_vox<<<BATCH * 128, 256, 0, stream>>>(F0, sums, sqs, g_pin, b_pin, coords, bbox, F1, cnt);
  k_voxnorm<<<(BATCH * CONN * NVOX) / 256, 256, 0, stream>>>(F1, cnt, H0);

  // conv3d #1 (WMMA implicit GEMM) + GN + SiLU
  k_conv3d<<<BATCH * 1024, 256, 0, stream>>>(Av1, H0, F2);
  k_stats<<<BATCH * CONN, 256, 0, stream>>>(F2, sums, sqs);
  k_gn_act<<<BATCH * 128, 256, 0, stream>>>(F2, sums, sqs, g_v1, b_v1, H1, nullptr);

  // conv3d #2 + GN + SiLU (f32 out for gather)
  k_conv3d<<<BATCH * 1024, 256, 0, stream>>>(Av2, H1, F2);
  k_stats<<<BATCH * CONN, 256, 0, stream>>>(F2, sums, sqs);
  k_gn_act<<<BATCH * 128, 256, 0, stream>>>(F2, sums, sqs, g_v2, b_v2, nullptr, F1);

  // devoxelize -> bf16, then fuse & skip GEMMs (WMMA)
  k_devox<<<BATCH * 128, 256, 0, stream>>>(F1, coords, bbox, H0);
  k_gemm<<<BATCH * 1024, 256, 0, stream>>>(Afuse, H0, F2, CONN);
  k_gemm<<<BATCH * 1024, 256, 0, stream>>>(Askip, Hf, F0, CINN);
  k_stats<<<BATCH * CONN, 256, 0, stream>>>(F2, sums, sqs);

  // AdaIN params + final fuse
  k_adain<<<1, 256, 0, stream>>>(style, w_as, b_as, w_ab, b_ab, sbuf, bbuf);
  k_final<<<BATCH * 128, 256, 0, stream>>>(F2, sums, sqs, g_f, b_f, F0, sbuf, bbuf, (float*)d_out);
}
